// PI05Pytorch_49761491091930
// MI455X (gfx1250) — compile-verified, matrix operand order checked by reading, not run on hardware
//
#include <hip/hip_runtime.h>

typedef __bf16 bf16_t;
typedef __attribute__((ext_vector_type(16))) __bf16 v16bf;
typedef __attribute__((ext_vector_type(8)))  __bf16 v8bf;
typedef __attribute__((ext_vector_type(8)))  float  v8f;

#define B_   2
#define P_   768
#define SX_  64
#define S_   832
#define D0_  2048
#define D1_  1024
#define DC_  1024
#define H_   8
#define DH_  256
#define F0_  16384
#define F1_  4096
#define EPS_ 1e-6f
#define SCALE_ 0.0625f   // DH^-0.5 = 1/16

#define LDT 40  // padded LDS row stride in bf16 elems (80B -> conflict-spread b128 reads)

// ---- CDNA5 async global->LDS (ASYNCcnt-tracked, no VGPR round trip) ----
__device__ __forceinline__ void g2l_async_b128(bf16_t* lds, const bf16_t* g) {
  uint32_t l = (uint32_t)(uintptr_t)lds;   // low 32 bits of generic LDS addr = LDS offset
  asm volatile("global_load_async_to_lds_b128 %0, %1, off"
               :: "v"(l), "v"(g) : "memory");
}
__device__ __forceinline__ void wait_async0() {
  asm volatile("s_wait_asynccnt 0x0" ::: "memory");
}

// ---------------------------------------------------------------------------
// Generic batched bf16 WMMA GEMM: C[M,N] = A[M,K] * B[K,N], fp32 accumulate.
// TRANSB=true: B element (k,n) read from Bm[n*ldb + k]  (used for q·k^T).
// Full interior tiles use a double-buffered async-to-LDS pipeline; edge tiles
// use a masked synchronous path. Batch z: z1=z/batch2, z2=z%batch2.
// ---------------------------------------------------------------------------
template<bool TRANSB>
__global__ __launch_bounds__(256) void gemm_bf16_kernel(
    const bf16_t* __restrict__ A, const bf16_t* __restrict__ Bm,
    float* __restrict__ C,
    int M, int N, int K, int lda, int ldb, int ldc,
    int batch2, long sA1, long sA2, long sB1, long sB2, long sC1, long sC2)
{
  __shared__ __align__(16) bf16_t Asm[2][128 * LDT];
  __shared__ __align__(16) bf16_t Bsm[2][128 * LDT];

  const int z  = blockIdx.z;
  const int z1 = z / batch2;
  const int z2 = z - z1 * batch2;
  A  += z1 * sA1 + z2 * sA2;
  Bm += z1 * sB1 + z2 * sB2;
  C  += z1 * sC1 + z2 * sC2;

  const int tid  = threadIdx.x;
  const int lane = tid & 31;
  const int wid  = tid >> 5;   // 0..7
  const int wm   = wid >> 1;   // 0..3 : 32-row slab
  const int wn   = wid & 1;    // 0..1 : 64-col slab
  const int row0 = blockIdx.x * 128;
  const int col0 = blockIdx.y * 128;
  const int lr   = lane & 15;
  const int hi   = lane >> 4;

  v8f acc[2][4];
#pragma unroll
  for (int mi = 0; mi < 2; mi++)
#pragma unroll
    for (int ni = 0; ni < 4; ni++)
#pragma unroll
      for (int j = 0; j < 8; j++) acc[mi][ni][j] = 0.f;

  // one 16x16x32 WMMA K-step per tile pair, fragments per ISA 16-bit layout
  auto compute_step = [&](const bf16_t* As, const bf16_t* Bs) {
    v16bf afrag[2], bfrag[4];
#pragma unroll
    for (int mi = 0; mi < 2; mi++) {
      const bf16_t* ap = As + (wm * 32 + mi * 16 + lr) * LDT;
      v8bf lo = *(const v8bf*)(ap + hi * 8);
      v8bf h8 = *(const v8bf*)(ap + 16 + hi * 8);
#pragma unroll
      for (int j = 0; j < 8; j++) { afrag[mi][j] = lo[j]; afrag[mi][j + 8] = h8[j]; }
    }
#pragma unroll
    for (int ni = 0; ni < 4; ni++) {
      const bf16_t* bp = Bs + (wn * 64 + ni * 16 + lr) * LDT;
      v8bf lo = *(const v8bf*)(bp + hi * 8);
      v8bf h8 = *(const v8bf*)(bp + 16 + hi * 8);
#pragma unroll
      for (int j = 0; j < 8; j++) { bfrag[ni][j] = lo[j]; bfrag[ni][j + 8] = h8[j]; }
    }
#pragma unroll
    for (int mi = 0; mi < 2; mi++)
#pragma unroll
      for (int ni = 0; ni < 4; ni++)
        acc[mi][ni] = __builtin_amdgcn_wmma_f32_16x16x32_bf16(
            false, afrag[mi], false, bfrag[ni], (short)0, acc[mi][ni], false, false);
  };

  const bool full = (row0 + 128 <= M) && (col0 + 128 <= N);
  if (full) {
    // ------------- double-buffered async pipeline (interior tiles) ----------
    auto issue = [&](int k0, int par) {
#pragma unroll
      for (int it = 0; it < 2; it++) {           // A tile: 512 x 16B chunks
        int s = tid + it * 256;
        int r = s >> 2, kk8 = (s & 3) * 8;
        g2l_async_b128(&Asm[par][r * LDT + kk8],
                       A + (long)(row0 + r) * lda + k0 + kk8);
      }
      if constexpr (TRANSB) {
#pragma unroll
        for (int it = 0; it < 2; it++) {         // B tile, k-contiguous rows
          int s = tid + it * 256;
          int n = s >> 2, kk8 = (s & 3) * 8;
          g2l_async_b128(&Bsm[par][n * LDT + kk8],
                         Bm + (long)(col0 + n) * ldb + k0 + kk8);
        }
      } else {
#pragma unroll
        for (int it = 0; it < 2; it++) {         // B tile, transpose via VGPRs
          int s = tid + it * 256;
          int kk = s >> 4, n8 = (s & 15) * 8;
          v8bf vv = *(const v8bf*)(Bm + (long)(k0 + kk) * ldb + col0 + n8);
#pragma unroll
          for (int j = 0; j < 8; j++) Bsm[par][(n8 + j) * LDT + kk] = vv[j];
        }
      }
    };

    issue(0, 0);
    const int nk = K >> 5;
    for (int kt = 0; kt < nk; kt++) {
      wait_async0();          // own async stores to LDS landed
      __syncthreads();        // everyone's stores landed (incl. DScnt for scatter)
      if (kt + 1 < nk) issue((kt + 1) << 5, (kt + 1) & 1);
      compute_step(Asm[kt & 1], Bsm[kt & 1]);
    }
  } else {
    // ------------------- masked synchronous path (edge tiles) ---------------
    for (int k0 = 0; k0 < K; k0 += 32) {
      __syncthreads();
#pragma unroll
      for (int it = 0; it < 2; it++) {
        int s = tid + it * 256;
        int r = s >> 2, kk8 = (s & 3) * 8;
        int gr = row0 + r;
        v8bf vv;
#pragma unroll
        for (int j = 0; j < 8; j++) vv[j] = (bf16_t)0.f;
        if (gr < M) {
          const bf16_t* src = A + (long)gr * lda + k0 + kk8;
          vv = *(const v8bf*)src;
          if (k0 + 32 < K) __builtin_prefetch(src + 32, 0, 1);
        }
        *(v8bf*)(&Asm[0][r * LDT + kk8]) = vv;
      }
      if constexpr (TRANSB) {
#pragma unroll
        for (int it = 0; it < 2; it++) {
          int s = tid + it * 256;
          int n = s >> 2, kk8 = (s & 3) * 8;
          int gn = col0 + n;
          v8bf vv;
#pragma unroll
          for (int j = 0; j < 8; j++) vv[j] = (bf16_t)0.f;
          if (gn < N) {
            const bf16_t* src = Bm + (long)gn * ldb + k0 + kk8;
            vv = *(const v8bf*)src;
            if (k0 + 32 < K) __builtin_prefetch(src + 32, 0, 1);
          }
          *(v8bf*)(&Bsm[0][n * LDT + kk8]) = vv;
        }
      } else {
#pragma unroll
        for (int it = 0; it < 2; it++) {
          int s = tid + it * 256;
          int kk = s >> 4, n8 = (s & 15) * 8;
          int gn = col0 + n8;
          v8bf vv;
#pragma unroll
          for (int j = 0; j < 8; j++) vv[j] = (bf16_t)0.f;
          if (gn < N) {
            const bf16_t* src = Bm + (long)(k0 + kk) * ldb + gn;
            vv = *(const v8bf*)src;
            if (k0 + 32 < K) __builtin_prefetch(src + (long)32 * ldb, 0, 1);
          }
#pragma unroll
          for (int j = 0; j < 8; j++) Bsm[0][(n8 + j) * LDT + kk] = vv[j];
        }
      }
      __syncthreads();
      compute_step(Asm[0], Bsm[0]);
    }
  }

  // ---- store: VGPR j of acc holds (M = hi*8+j, N = lr) of each 16x16 tile ----
#pragma unroll
  for (int mi = 0; mi < 2; mi++) {
#pragma unroll
    for (int ni = 0; ni < 4; ni++) {
      int c = col0 + wn * 64 + ni * 16 + lr;
      if (c < N) {
        int rbase = row0 + wm * 32 + mi * 16 + hi * 8;
#pragma unroll
        for (int j = 0; j < 8; j++) {
          int r = rbase + j;
          if (r < M) C[(long)r * ldc + c] = acc[mi][ni][j];
        }
      }
    }
  }
}

// ---------------------------------------------------------------------------
// fp32 -> bf16 conversion
// ---------------------------------------------------------------------------
__global__ void cvt_f32_bf16(const float* __restrict__ s, bf16_t* __restrict__ d, long n) {
  long i = ((long)blockIdx.x * blockDim.x + threadIdx.x) * 4;
#pragma unroll
  for (int j = 0; j < 4; j++)
    if (i + j < n) d[i + j] = (bf16_t)s[i + j];
}

// ---------------------------------------------------------------------------
// RMSNorm / AdaRMS -> bf16.  out = x * rsqrt(mean(x^2)+eps) * (1+w) [* (1+scale_b)]
// ---------------------------------------------------------------------------
__global__ __launch_bounds__(256) void rms_bf16(
    const float* __restrict__ x, const float* __restrict__ w,
    const float* __restrict__ scale, int ldscale,
    bf16_t* __restrict__ out, int d, int rowsPerB)
{
  int row = blockIdx.x;
  const float* xr = x + (long)row * d;
  __shared__ float red[256];
  int tid = threadIdx.x;
  float s = 0.f;
  for (int j = tid; j < d; j += 256) { float v = xr[j]; s += v * v; }
  red[tid] = s; __syncthreads();
  for (int off = 128; off > 0; off >>= 1) {
    if (tid < off) red[tid] += red[tid + off];
    __syncthreads();
  }
  float inv = rsqrtf(red[0] / (float)d + EPS_);
  const float* sc = scale ? scale + (long)(row / rowsPerB) * ldscale : nullptr;
  for (int j = tid; j < d; j += 256) {
    float v = xr[j] * inv * (1.f + w[j]);
    if (sc) v *= (1.f + sc[j]);
    out[(long)row * d + j] = (bf16_t)v;
  }
}

// ---------------------------------------------------------------------------
// RoPE + (q scale) + bf16 convert.  Unit h<H -> q head h; h==H -> k.
// ---------------------------------------------------------------------------
__global__ __launch_bounds__(128) void rope_qk(
    const float* __restrict__ qf, const float* __restrict__ kf,
    bf16_t* __restrict__ qb, bf16_t* __restrict__ kb)
{
  int u = blockIdx.x;
  int h = u % (H_ + 1);
  int t = u / (H_ + 1);
  int s = t % S_;
  int b = t / S_;
  int i = threadIdx.x; // 0..127
  float invf = __expf(-(float)i * (9.210340371976184f / 128.f)); // 10000^(-i/128)
  float ang = (float)s * invf;
  float c = cosf(ang), sn = sinf(ang);
  if (h < H_) {
    long base = ((long)b * S_ + s) * (H_ * DH_) + (long)h * DH_;
    float a1 = qf[base + i], a2 = qf[base + i + 128];
    qb[base + i]       = (bf16_t)((a1 * c - a2 * sn) * SCALE_);
    qb[base + i + 128] = (bf16_t)((a2 * c + a1 * sn) * SCALE_);
  } else {
    long base = ((long)b * S_ + s) * DH_;
    float a1 = kf[base + i], a2 = kf[base + i + 128];
    kb[base + i]       = (bf16_t)(a1 * c - a2 * sn);
    kb[base + i + 128] = (bf16_t)(a2 * c + a1 * sn);
  }
}

// ---------------------------------------------------------------------------
// Masked softmax row (prefix-LM mask recomputed analytically) -> bf16 probs.
// ---------------------------------------------------------------------------
__global__ __launch_bounds__(256) void softmax_bf16(
    const float* __restrict__ sc, bf16_t* __restrict__ pb)
{
  long blk = blockIdx.x;
  int i = (int)(blk % S_);
  long z = blk / S_;
  const float* row = sc + (z * S_ + i) * (long)S_;
  bf16_t* out = pb + (z * S_ + i) * (long)S_;
  __shared__ float red[256];
  int tid = threadIdx.x;
  float mx = -3.4e38f;
  for (int j = tid; j < S_; j += 256)
    if ((j < P_) || (j <= i)) mx = fmaxf(mx, row[j]);
  red[tid] = mx; __syncthreads();
  for (int off = 128; off > 0; off >>= 1) {
    if (tid < off) red[tid] = fmaxf(red[tid], red[tid + off]);
    __syncthreads();
  }
  mx = red[0]; __syncthreads();
  float sum = 0.f;
  for (int j = tid; j < S_; j += 256)
    if ((j < P_) || (j <= i)) sum += __expf(row[j] - mx);
  red[tid] = sum; __syncthreads();
  for (int off = 128; off > 0; off >>= 1) {
    if (tid < off) red[tid] += red[tid + off];
    __syncthreads();
  }
  float inv = 1.f / red[0];
  for (int j = tid; j < S_; j += 256) {
    bool m = (j < P_) || (j <= i);
    out[j] = (bf16_t)(m ? __expf(row[j] - mx) * inv : 0.f);
  }
}

// ---------------------------------------------------------------------------
// GeGLU: out = gelu_tanh(g) * u  -> bf16
// ---------------------------------------------------------------------------
__global__ void geglu_bf16(const float* __restrict__ g, const float* __restrict__ u,
                           bf16_t* __restrict__ o, long n) {
  long i = (long)blockIdx.x * blockDim.x + threadIdx.x;
  if (i < n) {
    float x = g[i];
    float t = tanhf(0.7978845608028654f * (x + 0.044715f * x * x * x));
    o[i] = (bf16_t)(0.5f * x * (1.f + t) * u[i]);
  }
}

__global__ void add2(const float* __restrict__ a, const float* __restrict__ b,
                     float* __restrict__ o, long n) {
  long i = (long)blockIdx.x * blockDim.x + threadIdx.x;
  if (i < n) o[i] = a[i] + b[i];
}

// o = a + gate[b_row, j] * y   (gate row stride ldg, raw gate, no 1+)
__global__ void add2_gated(const float* __restrict__ a, const float* __restrict__ y,
                           const float* __restrict__ g, int ldg, int rowsPerB, int d,
                           float* __restrict__ o, long n) {
  long i = (long)blockIdx.x * blockDim.x + threadIdx.x;
  if (i < n) {
    long r = i / d;
    int j = (int)(i - r * d);
    int b = (int)(r / rowsPerB);
    o[i] = a[i] + g[(long)b * ldg + j] * y[i];
  }
}

// ---------------------------------------------------------------------------
extern "C" void kernel_launch(void* const* d_in, const int* in_sizes, int n_in,
                              void* d_out, int out_size, void* d_ws, size_t ws_size,
                              hipStream_t stream)
{
  (void)in_sizes; (void)n_in; (void)out_size; (void)ws_size;

  const float* x0    = (const float*)d_in[0];
  const float* x1    = (const float*)d_in[1];
  const float* cond  = (const float*)d_in[2];
  const float* ln1w0 = (const float*)d_in[3];
  const float* wq0   = (const float*)d_in[4];
  const float* wk0   = (const float*)d_in[5];
  const float* wv0   = (const float*)d_in[6];
  const float* wo0   = (const float*)d_in[7];
  const float* ln2w0 = (const float*)d_in[8];
  const float* wg0   = (const float*)d_in[9];
  const float* wu0   = (const float*)d_in[10];
  const float* wd0   = (const float*)d_in[11];
  const float* ln1w1 = (const float*)d_in[12];
  const float* ada1  = (const float*)d_in[13];
  const float* wq1   = (const float*)d_in[14];
  const float* wk1   = (const float*)d_in[15];
  const float* wv1   = (const float*)d_in[16];
  const float* wo1   = (const float*)d_in[17];
  const float* ln2w1 = (const float*)d_in[18];
  const float* ada2  = (const float*)d_in[19];
  const float* wg1   = (const float*)d_in[20];
  const float* wu1   = (const float*)d_in[21];
  const float* wd1   = (const float*)d_in[22];
  // d_in[23] mask, d_in[24] position_ids: recomputed analytically in-kernel.

  float* out = (float*)d_out;

  char* cur = (char*)d_ws;
  auto alloc   = [&](size_t bytes) -> char* {
    char* p = cur; cur += (bytes + 255) & ~(size_t)255; return p;
  };
  auto allocBF = [&](long n) -> bf16_t* { return (bf16_t*)alloc((size_t)n * 2); };
  auto allocF  = [&](long n) -> float*  { return (float*)alloc((size_t)n * 4); };

  // bf16 staging
  bf16_t* wq0b  = allocBF((long)D0_ * H_ * DH_);
  bf16_t* wk0b  = allocBF((long)D0_ * DH_);
  bf16_t* wv0b  = allocBF((long)D0_ * DH_);
  bf16_t* wo0b  = allocBF((long)H_ * DH_ * D0_);
  bf16_t* wg0b  = allocBF((long)D0_ * F0_);
  bf16_t* wu0b  = allocBF((long)D0_ * F0_);
  bf16_t* wd0b  = allocBF((long)F0_ * D0_);
  bf16_t* wq1b  = allocBF((long)D1_ * H_ * DH_);
  bf16_t* wk1b  = allocBF((long)D1_ * DH_);
  bf16_t* wv1b  = allocBF((long)D1_ * DH_);
  bf16_t* wo1b  = allocBF((long)H_ * DH_ * D1_);
  bf16_t* wg1b  = allocBF((long)D1_ * F1_);
  bf16_t* wu1b  = allocBF((long)D1_ * F1_);
  bf16_t* wd1b  = allocBF((long)F1_ * D1_);
  bf16_t* ada1b = allocBF((long)DC_ * 2 * D1_);
  bf16_t* ada2b = allocBF((long)DC_ * 2 * D1_);
  bf16_t* condb = allocBF((long)B_ * DC_);
  bf16_t* h0b   = allocBF((long)B_ * P_ * D0_);
  bf16_t* h1b   = allocBF((long)B_ * SX_ * D1_);
  bf16_t* qb    = allocBF((long)B_ * S_ * H_ * DH_);
  bf16_t* kbuf  = allocBF((long)B_ * S_ * DH_);
  bf16_t* vbuf  = allocBF((long)B_ * S_ * DH_);
  bf16_t* probsb= allocBF((long)B_ * H_ * S_ * S_);
  bf16_t* attb  = allocBF((long)B_ * S_ * H_ * DH_);
  bf16_t* r0nb  = allocBF((long)B_ * P_ * D0_);
  bf16_t* n1b   = allocBF((long)B_ * SX_ * D1_);
  bf16_t* gel0b = allocBF((long)B_ * P_ * F0_);
  bf16_t* gel1b = allocBF((long)B_ * SX_ * F1_);

  // fp32 intermediates
  float* adao1 = allocF((long)B_ * 2 * D1_);
  float* adao2 = allocF((long)B_ * 2 * D1_);
  float* qf    = allocF((long)B_ * S_ * H_ * DH_);
  float* kf    = allocF((long)B_ * S_ * DH_);
  float* vf    = allocF((long)B_ * S_ * DH_);
  float* scrs  = allocF((long)B_ * H_ * S_ * S_);
  float* attf  = allocF((long)B_ * S_ * H_ * DH_);
  float* ap0   = allocF((long)B_ * P_ * D0_);
  float* ap1   = allocF((long)B_ * SX_ * D1_);
  float* r0    = allocF((long)B_ * P_ * D0_);
  float* r1    = allocF((long)B_ * SX_ * D1_);
  float* g0    = allocF((long)B_ * P_ * F0_);
  float* u0    = allocF((long)B_ * P_ * F0_);
  float* g1    = allocF((long)B_ * SX_ * F1_);
  float* u1    = allocF((long)B_ * SX_ * F1_);
  float* mlp0  = allocF((long)B_ * P_ * D0_);
  float* mlp1  = allocF((long)B_ * SX_ * D1_);

  auto cvt = [&](const float* s, bf16_t* d, long n) {
    int blocks = (int)(((n + 3) / 4 + 255) / 256);
    cvt_f32_bf16<<<dim3(blocks), dim3(256), 0, stream>>>(s, d, n);
  };
  auto gemm = [&](const bf16_t* Am, const bf16_t* Wm, float* Cm,
                  int M, int N, int K, int lda, int ldb, int ldc,
                  int nz, int b2, long sA1, long sA2, long sB1, long sB2,
                  long sC1, long sC2, bool transb) {
    dim3 g((M + 127) / 128, (N + 127) / 128, nz), blk(256);
    if (transb)
      gemm_bf16_kernel<true><<<g, blk, 0, stream>>>(Am, Wm, Cm, M, N, K, lda, ldb, ldc,
                                                    b2, sA1, sA2, sB1, sB2, sC1, sC2);
    else
      gemm_bf16_kernel<false><<<g, blk, 0, stream>>>(Am, Wm, Cm, M, N, K, lda, ldb, ldc,
                                                     b2, sA1, sA2, sB1, sB2, sC1, sC2);
  };

  // 1) weight + cond conversions
  cvt(wq0, wq0b, (long)D0_ * H_ * DH_);
  cvt(wk0, wk0b, (long)D0_ * DH_);
  cvt(wv0, wv0b, (long)D0_ * DH_);
  cvt(wo0, wo0b, (long)H_ * DH_ * D0_);
  cvt(wg0, wg0b, (long)D0_ * F0_);
  cvt(wu0, wu0b, (long)D0_ * F0_);
  cvt(wd0, wd0b, (long)F0_ * D0_);
  cvt(wq1, wq1b, (long)D1_ * H_ * DH_);
  cvt(wk1, wk1b, (long)D1_ * DH_);
  cvt(wv1, wv1b, (long)D1_ * DH_);
  cvt(wo1, wo1b, (long)H_ * DH_ * D1_);
  cvt(wg1, wg1b, (long)D1_ * F1_);
  cvt(wu1, wu1b, (long)D1_ * F1_);
  cvt(wd1, wd1b, (long)F1_ * D1_);
  cvt(ada1, ada1b, (long)DC_ * 2 * D1_);
  cvt(ada2, ada2b, (long)DC_ * 2 * D1_);
  cvt(cond, condb, (long)B_ * DC_);

  // 2) norms + AdaRMS conditioning
  rms_bf16<<<B_ * P_, 256, 0, stream>>>(x0, ln1w0, nullptr, 0, h0b, D0_, 1);
  gemm(condb, ada1b, adao1, B_, 2 * D1_, DC_, DC_, 2 * D1_, 2 * D1_,
       1, 1, 0, 0, 0, 0, 0, 0, false);
  gemm(condb, ada2b, adao2, B_, 2 * D1_, DC_, DC_, 2 * D1_, 2 * D1_,
       1, 1, 0, 0, 0, 0, 0, 0, false);
  rms_bf16<<<B_ * SX_, 256, 0, stream>>>(x1, ln1w1, adao1, 2 * D1_, h1b, D1_, SX_);

  // 3) QKV projections -> qf[B,S,H*DH], kf/vf[B,S,DH]
  gemm(h0b, wq0b, qf, P_, H_ * DH_, D0_, D0_, H_ * DH_, H_ * DH_,
       B_, 1, (long)P_ * D0_, 0, 0, 0, (long)S_ * H_ * DH_, 0, false);
  gemm(h0b, wk0b, kf, P_, DH_, D0_, D0_, DH_, DH_,
       B_, 1, (long)P_ * D0_, 0, 0, 0, (long)S_ * DH_, 0, false);
  gemm(h0b, wv0b, vf, P_, DH_, D0_, D0_, DH_, DH_,
       B_, 1, (long)P_ * D0_, 0, 0, 0, (long)S_ * DH_, 0, false);
  gemm(h1b, wq1b, qf + (long)P_ * H_ * DH_, SX_, H_ * DH_, D1_, D1_, H_ * DH_, H_ * DH_,
       B_, 1, (long)SX_ * D1_, 0, 0, 0, (long)S_ * H_ * DH_, 0, false);
  gemm(h1b, wk1b, kf + (long)P_ * DH_, SX_, DH_, D1_, D1_, DH_, DH_,
       B_, 1, (long)SX_ * D1_, 0, 0, 0, (long)S_ * DH_, 0, false);
  gemm(h1b, wv1b, vf + (long)P_ * DH_, SX_, DH_, D1_, D1_, DH_, DH_,
       B_, 1, (long)SX_ * D1_, 0, 0, 0, (long)S_ * DH_, 0, false);

  // 4) RoPE (+ q*SCALE) -> bf16; v -> bf16
  rope_qk<<<B_ * S_ * (H_ + 1), 128, 0, stream>>>(qf, kf, qb, kbuf);
  cvt(vf, vbuf, (long)B_ * S_ * DH_);

  // 5) scores = (q*SCALE) . k^T, batched over (b,h)
  gemm(qb, kbuf, scrs, S_, S_, DH_, H_ * DH_, DH_, S_,
       B_ * H_, H_,
       (long)S_ * H_ * DH_, (long)DH_,
       (long)S_ * DH_, 0,
       (long)H_ * S_ * S_, (long)S_ * S_, true);

  // 6) masked softmax -> bf16 probs
  softmax_bf16<<<B_ * H_ * S_, 256, 0, stream>>>(scrs, probsb);

  // 7) att = probs @ v -> attf[B,S,H*DH]
  gemm(probsb, vbuf, attf, S_, DH_, S_, S_, DH_, H_ * DH_,
       B_ * H_, H_,
       (long)H_ * S_ * S_, (long)S_ * S_,
       (long)S_ * DH_, 0,
       (long)S_ * H_ * DH_, (long)DH_, false);
  cvt(attf, attb, (long)B_ * S_ * H_ * DH_);

  // 8) output projections
  gemm(attb, wo0b, ap0, P_, D0_, H_ * DH_, H_ * DH_, D0_, D0_,
       B_, 1, (long)S_ * H_ * DH_, 0, 0, 0, (long)P_ * D0_, 0, false);
  gemm(attb + (long)P_ * H_ * DH_, wo1b, ap1, SX_, D1_, H_ * DH_, H_ * DH_, D1_, D1_,
       B_, 1, (long)S_ * H_ * DH_, 0, 0, 0, (long)SX_ * D1_, 0, false);

  // 9) residuals
  long n0 = (long)B_ * P_ * D0_;
  long n1 = (long)B_ * SX_ * D1_;
  add2<<<(int)((n0 + 255) / 256), 256, 0, stream>>>(x0, ap0, r0, n0);
  add2_gated<<<(int)((n1 + 255) / 256), 256, 0, stream>>>(
      x1, ap1, adao1 + D1_, 2 * D1_, SX_, D1_, r1, n1);

  // 10) second norms
  rms_bf16<<<B_ * P_, 256, 0, stream>>>(r0, ln2w0, nullptr, 0, r0nb, D0_, 1);
  rms_bf16<<<B_ * SX_, 256, 0, stream>>>(r1, ln2w1, adao2, 2 * D1_, n1b, D1_, SX_);

  // 11) MLP stream 0
  gemm(r0nb, wg0b, g0, P_, F0_, D0_, D0_, F0_, F0_,
       B_, 1, (long)P_ * D0_, 0, 0, 0, (long)P_ * F0_, 0, false);
  gemm(r0nb, wu0b, u0, P_, F0_, D0_, D0_, F0_, F0_,
       B_, 1, (long)P_ * D0_, 0, 0, 0, (long)P_ * F0_, 0, false);
  long ng0 = (long)B_ * P_ * F0_;
  geglu_bf16<<<(int)((ng0 + 255) / 256), 256, 0, stream>>>(g0, u0, gel0b, ng0);
  gemm(gel0b, wd0b, mlp0, P_, D0_, F0_, F0_, D0_, D0_,
       B_, 1, (long)P_ * F0_, 0, 0, 0, (long)P_ * D0_, 0, false);
  add2<<<(int)((n0 + 255) / 256), 256, 0, stream>>>(r0, mlp0, out, n0);

  // 12) MLP stream 1 (gated)
  gemm(n1b, wg1b, g1, SX_, F1_, D1_, D1_, F1_, F1_,
       B_, 1, (long)SX_ * D1_, 0, 0, 0, (long)SX_ * F1_, 0, false);
  gemm(n1b, wu1b, u1, SX_, F1_, D1_, D1_, F1_, F1_,
       B_, 1, (long)SX_ * D1_, 0, 0, 0, (long)SX_ * F1_, 0, false);
  long ng1 = (long)B_ * SX_ * F1_;
  geglu_bf16<<<(int)((ng1 + 255) / 256), 256, 0, stream>>>(g1, u1, gel1b, ng1);
  gemm(gel1b, wd1b, mlp1, SX_, D1_, F1_, F1_, D1_, D1_,
       B_, 1, (long)SX_ * F1_, 0, 0, 0, (long)SX_ * D1_, 0, false);
  add2_gated<<<(int)((n1 + 255) / 256), 256, 0, stream>>>(
      r1, mlp1, adao2 + D1_, 2 * D1_, SX_, D1_, out + n0, n1);
}
